// BezierDeformableAttention_44470091382917
// MI455X (gfx1250) — compile-verified, matrix operand order checked by reading, not run on hardware
//
#include <hip/hip_runtime.h>
#include <stdint.h>

// Problem constants (match reference)
#define Bq    4
#define Nq    900
#define Dd    256
#define Cc    256
#define Hh    200
#define Ww    200
#define HEADS 8
#define PTS   4
#define KK    10
#define HWsz  (Hh * Ww)          // 40000
#define BN    (Bq * Nq)          // 3600
#define BHW   (Bq * HWsz)        // 160000

typedef __attribute__((ext_vector_type(16))) __bf16          v16bf;
typedef __attribute__((ext_vector_type(2)))  __bf16          v2bf;
typedef __attribute__((ext_vector_type(8)))  float           v8f;
typedef __attribute__((ext_vector_type(2)))  float           v2f;
typedef __attribute__((ext_vector_type(8)))  unsigned short  v8us;
typedef __attribute__((ext_vector_type(16))) unsigned short  v16us;

// Native f32 -> bf16 conversions: let the compiler pick the hardware cvt
// (v_cvt_pk_bf16_f32 / v_cvt_bf16_f32 if gfx1250 has it; inline RNE otherwise).
__device__ __forceinline__ unsigned int f2bf_pk(float lo, float hi) {
  v2bf v = { (__bf16)lo, (__bf16)hi };
  return __builtin_bit_cast(unsigned int, v);
}
__device__ __forceinline__ unsigned short f2bf(float f) {
  return __builtin_bit_cast(unsigned short, (__bf16)f);
}
__device__ __forceinline__ float bf2f(unsigned short h) {
  unsigned int u = ((unsigned int)h) << 16;
  return __builtin_bit_cast(float, u);
}

// ---------------------------------------------------------------------------
// f32 -> bf16 conversion, 2-wide (grid-stride); n must be even (all sizes are)
// ---------------------------------------------------------------------------
__global__ void cvt_bf16_kernel(const float* __restrict__ src,
                                unsigned int* __restrict__ dst, int n2) {
  int i = blockIdx.x * blockDim.x + threadIdx.x;
  int stride = gridDim.x * blockDim.x;
  for (; i < n2; i += stride) {
    v2f v = ((const v2f*)src)[i];
    dst[i] = f2bf_pk(v.x, v.y);
  }
}

// ---------------------------------------------------------------------------
// bf16 WMMA GEMM:  out = A(MxK) * B(KxN) + bias [+ residual]
//   mode 0: A = bf16 row-major (M x Kd)
//   mode 1: A = f32, batch-transposed (bev): A[m,k] = Atf[b*Kd*HWsz + k*HWsz + p]
// Block tile 128x64, 8 waves: wave = 32 rows x 32 cols, 4 accumulators.
// Per k-tile (K=32): 8 x ds_load_b128 fragment loads -> 4 x v_wmma.
// ---------------------------------------------------------------------------
__global__ __launch_bounds__(256) void wmma_gemm_kernel(
    const unsigned short* __restrict__ Abf,
    const float*          __restrict__ Atf,
    int mode,
    const unsigned short* __restrict__ Bw,    // Kd x Nn row-major bf16
    const float*          __restrict__ bias,  // Nn
    const float*          __restrict__ residual, // M x Nn or null
    float*                __restrict__ outF,     // M x Nn or null
    unsigned short*       __restrict__ outH,     // M x Nn or null
    int M, int Nn, int Kd)
{
  __shared__ alignas(16) unsigned short As[128][32];  // [m][k]  (8 KB)
  __shared__ alignas(16) unsigned short Bs[64][32];   // transposed: [n][k] (4 KB)

  const int tid  = threadIdx.x;
  const int lane = tid & 31;
  const int wid  = tid >> 5;
  const int wr   = wid & 3;              // row 32-tile within block (0..3)
  const int wc   = wid >> 2;             // col 32-half within block (0..1)
  const int m0   = blockIdx.y * 128;
  const int n0   = blockIdx.x * 64;

  v8f acc[2][2];
  #pragma unroll
  for (int i = 0; i < 2; ++i)
    #pragma unroll
    for (int j = 0; j < 2; ++j)
      acc[i][j] = v8f{0.f,0.f,0.f,0.f,0.f,0.f,0.f,0.f};

  const int h   = lane >> 4;
  const int l15 = lane & 15;

  for (int kb = 0; kb < Kd; kb += 32) {
    // ---- stage A tile into LDS ----
    if (mode == 0) {
      // straight b128 copies: row = tid>>1 (128 rows), k-chunk = (tid&1)*16
      int row = tid >> 1;
      int k0  = (tid & 1) * 16;
      int m   = m0 + row;
      v8us z  = {0,0,0,0,0,0,0,0};
      if (m < M) {
        const v8us* src = (const v8us*)(Abf + (size_t)m * Kd + kb + k0);
        if (kb + 32 < Kd) __builtin_prefetch((const void*)(src + 2), 0, 3);
        ((v8us*)&As[row][k0])[0] = src[0];
        ((v8us*)&As[row][k0])[1] = src[1];
      } else {
        ((v8us*)&As[row][k0])[0] = z;
        ((v8us*)&As[row][k0])[1] = z;
      }
    } else {
      // bev: k-pair per thread, 8 consecutive m; pack bf16 pairs -> b32 stores
      int pk  = tid >> 4;              // k-pair index 0..15 -> k = 2pk, 2pk+1
      int mch = (tid & 15) * 8;        // m chunk (8 rows)
      int mbase = m0 + mch;
      int bb = mbase / HWsz;
      int pp = mbase - bb * HWsz;
      const float* s0 = Atf + ((size_t)bb * Kd + (kb + 2 * pk)) * HWsz + pp;
      const float* s1 = s0 + HWsz;
      if (kb + 32 < Kd) __builtin_prefetch((const void*)(s0 + (size_t)32 * HWsz), 0, 3);
      if (pp + 8 <= HWsz) {            // fast path: contiguous within batch
        #pragma unroll
        for (int i = 0; i < 8; ++i)
          *(unsigned int*)&As[mch + i][2 * pk] = f2bf_pk(s0[i], s1[i]);
      } else {                         // rare: tile crosses batch boundary
        #pragma unroll
        for (int i = 0; i < 8; ++i) {
          int m  = mbase + i;
          int b2 = m / HWsz;
          int p2 = m - b2 * HWsz;
          const float* t0 = Atf + ((size_t)b2 * Kd + (kb + 2 * pk)) * HWsz + p2;
          *(unsigned int*)&As[mch + i][2 * pk] = f2bf_pk(t0[0], t0[HWsz]);
        }
      }
    }
    // ---- stage B tile into LDS, transposed [n][k], packed b32 stores ----
    {
      int pk  = tid >> 4;              // k-pair 0..15
      int nch = (tid & 15) * 4;        // 4 columns
      const unsigned short* r0 = Bw + (size_t)(kb + 2 * pk) * Nn;
      const unsigned short* r1 = r0 + Nn;
      #pragma unroll
      for (int i = 0; i < 4; ++i) {
        int n = n0 + nch + i;
        unsigned int v = 0;
        if (n < Nn) v = (unsigned int)r0[n] | ((unsigned int)r1[n] << 16);
        *(unsigned int*)&Bs[nch + i][2 * pk] = v;
      }
    }
    __syncthreads();

    // ---- assemble fragments (two 128-bit LDS reads each) ----
    v16bf a[2], b[2];
    #pragma unroll
    for (int i = 0; i < 2; ++i) {
      const v8us* Ar = (const v8us*)&As[wr * 32 + i * 16 + l15][0];
      v16us a_us = __builtin_shufflevector(Ar[h], Ar[2 + h],
          0,1,2,3,4,5,6,7,8,9,10,11,12,13,14,15);
      a[i] = __builtin_bit_cast(v16bf, a_us);
    }
    #pragma unroll
    for (int j = 0; j < 2; ++j) {
      const v8us* Br = (const v8us*)&Bs[wc * 32 + j * 16 + l15][0];
      v16us b_us = __builtin_shufflevector(Br[h], Br[2 + h],
          0,1,2,3,4,5,6,7,8,9,10,11,12,13,14,15);
      b[j] = __builtin_bit_cast(v16bf, b_us);
    }

    #pragma unroll
    for (int i = 0; i < 2; ++i)
      #pragma unroll
      for (int j = 0; j < 2; ++j)
        acc[i][j] = __builtin_amdgcn_wmma_f32_16x16x32_bf16(
            false, a[i], false, b[j], (short)0, acc[i][j], false, false);
    __syncthreads();
  }

  // ---- epilogue: bias, residual, f32 + bf16 stores ----
  #pragma unroll
  for (int ti = 0; ti < 2; ++ti) {
    #pragma unroll
    for (int r = 0; r < 8; ++r) {
      int m = m0 + wr * 32 + ti * 16 + h * 8 + r;
      if (m >= M) continue;
      #pragma unroll
      for (int tj = 0; tj < 2; ++tj) {
        int n = n0 + wc * 32 + tj * 16 + l15;
        if (n < Nn) {
          float val = acc[ti][tj][r] + bias[n];
          if (residual) val += residual[(size_t)m * Nn + n];
          if (outF) outF[(size_t)m * Nn + n] = val;
          if (outH) outH[(size_t)m * Nn + n] = f2bf(val);
        }
      }
    }
  }
}

// ---------------------------------------------------------------------------
// Bezier + softmax + bilinear deformable sampling + head reduction.
// 1 block per (b,n) query; wave = head; lane = channel (hd = 32).
// ---------------------------------------------------------------------------
__global__ __launch_bounds__(256) void sample_kernel(
    const float* __restrict__ ctrl,         // (BN, 4, 2)
    const float* __restrict__ off,          // (BN, 64)
    const float* __restrict__ attn,         // (BN, 32)
    const unsigned short* __restrict__ vbf, // (BHW, 256) bf16
    const float* __restrict__ pcr,          // 6
    const int*   __restrict__ ss,           // (1,2) = {H, W}
    float* __restrict__ red,                // (BN, 256)
    unsigned short* __restrict__ redH)      // (BN, 256) bf16
{
  const int bn    = blockIdx.x;
  const int tid   = threadIdx.x;
  const int hHead = tid >> 5;
  const int ch    = tid & 31;
  const int b     = bn / Nq;

  const float rx0 = pcr[0], ry0 = pcr[1], rx1 = pcr[3], ry1 = pcr[4];
  const float invX = 1.0f / (rx1 - rx0);
  const float invY = 1.0f / (ry1 - ry0);
  const float normW = (float)ss[1];
  const float normH = (float)ss[0];

  float cpx[4], cpy[4];
  #pragma unroll
  for (int i = 0; i < 4; ++i) {
    cpx[i] = ctrl[(size_t)bn * 8 + i * 2 + 0];
    cpy[i] = ctrl[(size_t)bn * 8 + i * 2 + 1];
  }

  // softmax over the 4 points of this head
  float lg[4];
  #pragma unroll
  for (int p = 0; p < 4; ++p) lg[p] = attn[(size_t)bn * 32 + hHead * 4 + p];
  float mx = fmaxf(fmaxf(lg[0], lg[1]), fmaxf(lg[2], lg[3]));
  float wp[4], wsum = 0.f;
  #pragma unroll
  for (int p = 0; p < 4; ++p) { wp[p] = __expf(lg[p] - mx); wsum += wp[p]; }
  float invs = 1.0f / wsum;
  #pragma unroll
  for (int p = 0; p < 4; ++p) wp[p] *= invs;

  // normalized per-point offsets
  float ofx[4], ofy[4];
  #pragma unroll
  for (int p = 0; p < 4; ++p) {
    ofx[p] = off[(size_t)bn * 64 + hHead * 8 + p * 2 + 0] / normW;
    ofy[p] = off[(size_t)bn * 64 + hHead * 8 + p * 2 + 1] / normH;
  }

  const unsigned short* vbase = vbf + (size_t)b * HWsz * 256 + hHead * 32 + ch;

  auto corner = [&](int cx, int cy, float wgt, float& s) {
    bool valid = (cx >= 0) && (cx < Ww) && (cy >= 0) && (cy < Hh);
    int cxx = cx < 0 ? 0 : (cx > Ww - 1 ? Ww - 1 : cx);
    int cyy = cy < 0 ? 0 : (cy > Hh - 1 ? Hh - 1 : cy);
    float val = bf2f(vbase[(size_t)(cyy * Ww + cxx) * 256]);
    s += valid ? wgt * val : 0.f;
  };

  float acc = 0.f;
  for (int kk = 0; kk < KK; ++kk) {
    float t  = (float)kk * (1.0f / (KK - 1));
    float u  = 1.0f - t;
    float b0 = u * u * u;
    float b1 = 3.f * u * u * t;
    float b2 = 3.f * u * t * t;
    float b3 = t * t * t;
    float px = b0 * cpx[0] + b1 * cpx[1] + b2 * cpx[2] + b3 * cpx[3];
    float py = b0 * cpy[0] + b1 * cpy[1] + b2 * cpy[2] + b3 * cpy[3];
    float nx = (px - rx0) * invX;
    float ny = (py - ry0) * invY;
    nx = fminf(fmaxf(nx, 0.01f), 0.99f);
    ny = fminf(fmaxf(ny, 0.01f), 0.99f);

    #pragma unroll
    for (int p = 0; p < 4; ++p) {
      float gx = (nx + ofx[p]) * (float)Ww - 0.5f;
      float gy = (ny + ofy[p]) * (float)Hh - 0.5f;
      float fx = floorf(gx), fy = floorf(gy);
      int ix = (int)fx, iy = (int)fy;
      float wx = gx - fx, wy = gy - fy;
      float s = 0.f;
      corner(ix    , iy    , (1.f - wx) * (1.f - wy), s);
      corner(ix + 1, iy    ,         wx * (1.f - wy), s);
      corner(ix    , iy + 1, (1.f - wx) *         wy, s);
      corner(ix + 1, iy + 1,         wx *         wy, s);
      acc += wp[p] * s;
    }
  }
  size_t o = (size_t)bn * 256 + hHead * 32 + ch;
  red[o]  = acc;
  redH[o] = f2bf(acc);
}

// ---------------------------------------------------------------------------
// Host-side launch
// ---------------------------------------------------------------------------
extern "C" void kernel_launch(void* const* d_in, const int* in_sizes, int n_in,
                              void* d_out, int out_size, void* d_ws, size_t ws_size,
                              hipStream_t stream) {
  (void)in_sizes; (void)n_in; (void)out_size; (void)ws_size;

  const float* query = (const float*)d_in[0];
  const float* ctrl  = (const float*)d_in[1];
  const float* bev   = (const float*)d_in[2];
  const int*   ss    = (const int*)  d_in[3];
  const float* pcr   = (const float*)d_in[4];
  const float* Wq    = (const float*)d_in[5];
  const float* bq    = (const float*)d_in[6];
  const float* Wv    = (const float*)d_in[7];
  const float* bv    = (const float*)d_in[8];
  const float* Woff  = (const float*)d_in[9];
  const float* boff  = (const float*)d_in[10];
  const float* Wattn = (const float*)d_in[11];
  const float* battn = (const float*)d_in[12];
  const float* Wmo   = (const float*)d_in[13];
  const float* bmo   = (const float*)d_in[14];
  const float* Wo    = (const float*)d_in[15];
  const float* bo    = (const float*)d_in[16];
  float* out = (float*)d_out;

  char* ws = (char*)d_ws;
  auto alloc = [&](size_t bytes) -> char* {
    char* p = ws;
    ws += (bytes + 255) & ~(size_t)255;
    return p;
  };

  unsigned short* query_bf = (unsigned short*)alloc((size_t)BN * Dd * 2);
  unsigned short* Wq_bf    = (unsigned short*)alloc((size_t)Dd * Dd * 2);
  unsigned short* Wv_bf    = (unsigned short*)alloc((size_t)Cc * Dd * 2);
  unsigned short* Woff_bf  = (unsigned short*)alloc((size_t)Dd * 64 * 2);
  unsigned short* Wattn_bf = (unsigned short*)alloc((size_t)Dd * 32 * 2);
  unsigned short* Wmo_bf   = (unsigned short*)alloc((size_t)Dd * Dd * 2);
  unsigned short* Wo_bf    = (unsigned short*)alloc((size_t)Dd * Dd * 2);
  float*          q_f32    = (float*)         alloc((size_t)BN * Dd * 4);
  unsigned short* q_bf     = (unsigned short*)alloc((size_t)BN * Dd * 2);
  float*          off_f32  = (float*)         alloc((size_t)BN * 64 * 4);
  float*          attn_f32 = (float*)         alloc((size_t)BN * 32 * 4);
  unsigned short* v_bf     = (unsigned short*)alloc((size_t)BHW * Dd * 2);
  float*          red_f32  = (float*)         alloc((size_t)BN * Dd * 4);
  unsigned short* red_bf   = (unsigned short*)alloc((size_t)BN * Dd * 2);
  float*          o1_f32   = (float*)         alloc((size_t)BN * Dd * 4);
  unsigned short* o1_bf    = (unsigned short*)alloc((size_t)BN * Dd * 2);

  auto cvt = [&](const float* s, unsigned short* d, int n) {
    int n2 = n / 2;
    int blocks = (n2 + 255) / 256;
    if (blocks > 2048) blocks = 2048;
    cvt_bf16_kernel<<<blocks, 256, 0, stream>>>(s, (unsigned int*)d, n2);
  };
  cvt(query, query_bf, BN * Dd);
  cvt(Wq,    Wq_bf,    Dd * Dd);
  cvt(Wv,    Wv_bf,    Cc * Dd);
  cvt(Woff,  Woff_bf,  Dd * 64);
  cvt(Wattn, Wattn_bf, Dd * 32);
  cvt(Wmo,   Wmo_bf,   Dd * Dd);
  cvt(Wo,    Wo_bf,    Dd * Dd);

  const int mt = (BN + 127) / 128;   // 29 row tiles for the query-side GEMMs

  // q = query @ Wq + bq
  wmma_gemm_kernel<<<dim3(Dd / 64, mt), 256, 0, stream>>>(
      query_bf, nullptr, 0, Wq_bf, bq, nullptr, q_f32, q_bf, BN, Dd, Dd);

  // off = q @ Woff + boff
  wmma_gemm_kernel<<<dim3(1, mt), 256, 0, stream>>>(
      q_bf, nullptr, 0, Woff_bf, boff, nullptr, off_f32, nullptr, BN, 64, Dd);

  // attn_logits = q @ Wattn + battn
  wmma_gemm_kernel<<<dim3(1, mt), 256, 0, stream>>>(
      q_bf, nullptr, 0, Wattn_bf, battn, nullptr, attn_f32, nullptr, BN, 32, Dd);

  // v = bev^T @ Wv + bv   (transpose + f32->bf16 fused into LDS staging)
  wmma_gemm_kernel<<<dim3(Dd / 64, BHW / 128), 256, 0, stream>>>(
      nullptr, bev, 1, Wv_bf, bv, nullptr, nullptr, v_bf, BHW, Dd, Cc);

  // deformable sampling + head reduction
  sample_kernel<<<dim3(BN), 256, 0, stream>>>(
      ctrl, off_f32, attn_f32, v_bf, pcr, ss, red_f32, red_bf);

  // out1 = red @ Wmo + bmo + q
  wmma_gemm_kernel<<<dim3(Dd / 64, mt), 256, 0, stream>>>(
      red_bf, nullptr, 0, Wmo_bf, bmo, q_f32, o1_f32, o1_bf, BN, Dd, Dd);

  // out = out1 @ Wo + bo
  wmma_gemm_kernel<<<dim3(Dd / 64, mt), 256, 0, stream>>>(
      o1_bf, nullptr, 0, Wo_bf, bo, nullptr, out, nullptr, BN, Dd, Dd);
}
